// SpanSelection_54975581389327
// MI455X (gfx1250) — compile-verified, matrix operand order checked by reading, not run on hardware
//
#include <hip/hip_runtime.h>
#include <hip/hip_bf16.h>

// ---------------- types for CDNA5 WMMA ----------------
typedef __bf16 bf16_t;
typedef __attribute__((ext_vector_type(8)))  __bf16 v8bf;
typedef __attribute__((ext_vector_type(16))) __bf16 v16bf;
typedef __attribute__((ext_vector_type(8)))  float  v8f;
typedef int v4i_t __attribute__((vector_size(16)));   // matches builtin param type

// problem constants
#define BB 8
#define LL 256
#define DD 1024
#define AA 64
#define VV 128
#define EE 256
#define GG 2048
#define WW 255
#define D1 1025          // D+1
#define KPAD 1056        // D+1 padded to mult of 32 (33*32)
#define NPAD 1088        // T's N dim padded to mult of 64 (17*64)
#define NEGV (-1000.0f)

#define BSTRIDE 40       // padded LDS column stride (elements) to spread banks

// gfx1250 async global->LDS path (ASYNCcnt), with portable fallback
#if defined(__HIP_DEVICE_COMPILE__) && \
    __has_builtin(__builtin_amdgcn_global_load_async_to_lds_b128) && \
    __has_builtin(__builtin_amdgcn_s_wait_asynccnt)
#define USE_ASYNC_LDS 1
#endif

// ---------------- helpers ----------------
__device__ __forceinline__ v16bf load_frag(const bf16_t* __restrict__ p) {
  // 16-bit A/B fragment for v_wmma_*_16x16x32: per-lane 16 bf16 values in two
  // contiguous 8-element (16B) chunks at +0 and +16 elements.
  v8bf lo = *reinterpret_cast<const v8bf*>(p);
  v8bf hi = *reinterpret_cast<const v8bf*>(p + 16);
  v16bf r;
#pragma unroll
  for (int i = 0; i < 8; ++i) { r[i] = lo[i]; r[i + 8] = hi[i]; }
  return r;
}

// stage one 64(cols) x 32(k) bf16 B-tile into LDS; 128 threads, 2x16B each
__device__ __forceinline__ void fill_btile(const bf16_t* __restrict__ Bb, long ldb,
                                           int n0, int kb, bf16_t* __restrict__ dst,
                                           int tid) {
#pragma unroll
  for (int i = 0; i < 2; ++i) {
    int c    = tid + (i << 7);     // chunk id 0..255
    int colc = c >> 2;             // 0..63
    int ko   = (c & 3) << 3;       // 0,8,16,24 (elements)
    const bf16_t* src = Bb + (long)(n0 + colc) * ldb + kb + ko;
    bf16_t*       d   = dst + colc * BSTRIDE + ko;
#ifdef USE_ASYNC_LDS
    __builtin_amdgcn_global_load_async_to_lds_b128((v4i_t*)src, (v4i_t*)d, 0, 0);
#else
    *reinterpret_cast<v8bf*>(d) = *reinterpret_cast<const v8bf*>(src);
#endif
  }
}

// ---------------- generic bf16 WMMA GEMM:  C = A(MxK) * Bt(NxK)^T + bias ----
// 128-thread block (4 waves): 64(M) x 64(N) C tile; B k-slices staged in LDS
// (double-buffered), A fragments stream from global with register reuse.
// grid = (N/64, M/64, batch)
__global__ __launch_bounds__(128)
void k_gemm_bf16(const bf16_t* __restrict__ A, long lda, long strideA,
                 const bf16_t* __restrict__ Bt, long ldb, long strideB,
                 void* __restrict__ C, long ldc, long strideC,
                 const float* __restrict__ bias, int K, int out_bf16) {
  __shared__ bf16_t Bs[2][64 * BSTRIDE];

  const int tid  = threadIdx.x;
  const int wave = tid >> 5;
  const int lane = tid & 31;
  const int m0   = (blockIdx.y << 6) + (wave << 4);
  const int n0   = blockIdx.x << 6;
  const int b    = blockIdx.z;
  const int col  = lane & 15;
  const int koff = (lane >> 4) << 3;   // lanes 0-15: K 0-7/16-23 ; lanes 16-31: K 8-15/24-31

  const bf16_t* Ap = A  + (long)b * strideA + (long)(m0 + col) * lda + koff;
  const bf16_t* Bb = Bt + (long)b * strideB;

  v8f acc0 = {}, acc1 = {}, acc2 = {}, acc3 = {};

  fill_btile(Bb, ldb, n0, 0, Bs[0], tid);
  int buf = 0;

  for (int kb = 0; kb < K; kb += 32) {
#ifdef USE_ASYNC_LDS
    __builtin_amdgcn_s_wait_asynccnt(0);
#endif
    __syncthreads();
    if (kb + 32 < K) {
      fill_btile(Bb, ldb, n0, kb + 32, Bs[buf ^ 1], tid);
      __builtin_prefetch(Ap + kb + 32, 0, 3);
    }
    const bf16_t* bs = Bs[buf];
    v16bf af = load_frag(Ap + kb);
    v16bf b0 = load_frag(bs + ( 0 + col) * BSTRIDE + koff);
    v16bf b1 = load_frag(bs + (16 + col) * BSTRIDE + koff);
    v16bf b2 = load_frag(bs + (32 + col) * BSTRIDE + koff);
    v16bf b3 = load_frag(bs + (48 + col) * BSTRIDE + koff);
    acc0 = __builtin_amdgcn_wmma_f32_16x16x32_bf16(false, af, false, b0, (short)0, acc0, false, false);
    acc1 = __builtin_amdgcn_wmma_f32_16x16x32_bf16(false, af, false, b1, (short)0, acc1, false, false);
    acc2 = __builtin_amdgcn_wmma_f32_16x16x32_bf16(false, af, false, b2, (short)0, acc2, false, false);
    acc3 = __builtin_amdgcn_wmma_f32_16x16x32_bf16(false, af, false, b3, (short)0, acc3, false, false);
    buf ^= 1;
  }

  // C/D layout: VGPR r -> row m0 + r (+8 for lanes 16-31), col n0 + (lane&15)
  const int rbase = m0 + ((lane >> 4) << 3);
  float bi0 = bias ? bias[n0 +  0 + col] : 0.f;
  float bi1 = bias ? bias[n0 + 16 + col] : 0.f;
  float bi2 = bias ? bias[n0 + 32 + col] : 0.f;
  float bi3 = bias ? bias[n0 + 48 + col] : 0.f;

  if (out_bf16) {
    bf16_t* Ch = reinterpret_cast<bf16_t*>(C) + (long)b * strideC;
#pragma unroll
    for (int r = 0; r < 8; ++r) {
      long row = rbase + r;
      Ch[row * ldc + n0 +  0 + col] = (bf16_t)(acc0[r] + bi0);
      Ch[row * ldc + n0 + 16 + col] = (bf16_t)(acc1[r] + bi1);
      Ch[row * ldc + n0 + 32 + col] = (bf16_t)(acc2[r] + bi2);
      Ch[row * ldc + n0 + 48 + col] = (bf16_t)(acc3[r] + bi3);
    }
  } else {
    float* Cf = reinterpret_cast<float*>(C) + (long)b * strideC;
#pragma unroll
    for (int r = 0; r < 8; ++r) {
      long row = rbase + r;
      Cf[row * ldc + n0 +  0 + col] = acc0[r] + bi0;
      Cf[row * ldc + n0 + 16 + col] = acc1[r] + bi1;
      Cf[row * ldc + n0 + 32 + col] = acc2[r] + bi2;
      Cf[row * ldc + n0 + 48 + col] = acc3[r] + bi3;
    }
  }
}

// ---------------- data preparation kernels ----------------
__global__ void k_build_xb(const float* __restrict__ reprs, bf16_t* __restrict__ xb, int n) {
  int idx = blockIdx.x * blockDim.x + threadIdx.x;
  if (idx >= n) return;
  int k  = idx % KPAD;
  int bl = idx / KPAD;                 // b*L + l
  float v = (k < DD) ? reprs[(long)bl * DD + k] : (k == DD ? 1.0f : 0.0f);
  xb[idx] = (bf16_t)v;
}

__global__ void k_build_wt(const float* __restrict__ Wb, bf16_t* __restrict__ Wt, int n) {
  int idx = blockIdx.x * blockDim.x + threadIdx.x;
  if (idx >= n) return;
  int k = idx % KPAD;                  // input (i) dim of W
  int j = idx / KPAD;                  // output (j) dim of W
  float v = (j < D1 && k < D1) ? Wb[(long)k * D1 + j] : 0.0f;
  Wt[idx] = (bf16_t)v;
}

__global__ void k_f32_to_bf16(const float* __restrict__ s, bf16_t* __restrict__ d, int n) {
  int i = blockIdx.x * blockDim.x + threadIdx.x;
  if (i < n) d[i] = (bf16_t)s[i];
}

__global__ void k_zero_u32(unsigned int* __restrict__ p, int n) {
  int i = blockIdx.x * blockDim.x + threadIdx.x;
  if (i < n) p[i] = 0u;
}

__global__ void k_scatter_gold(const int* __restrict__ gs, unsigned char* __restrict__ gold) {
  int i = blockIdx.x * blockDim.x + threadIdx.x;
  if (i >= GG) return;
  int b = gs[3 * i + 0], l = gs[3 * i + 1], r = gs[3 * i + 2];
  gold[((long)b * LL + l) * LL + r] = 1;
}

// ---------------- span BCE loss (masked, per-block partials) ----------------
__global__ void k_span_loss(const float* __restrict__ S, const unsigned char* __restrict__ gold,
                            const int* __restrict__ seq_len, float* __restrict__ part) {
  __shared__ float red[256];
  int t = threadIdx.x;
  long idx = (long)blockIdx.x * 256 + t;      // over B*L*L = 524288
  int b = (int)(idx >> 16);
  int rem = (int)(idx & 65535);
  int x = rem >> 8, y = rem & 255;
  float v = 0.0f;
  if (x < y && y <= seq_len[b]) {
    float s = S[idx];
    float sp = fmaxf(s, 0.0f) + log1pf(expf(-fabsf(s)));   // stable softplus
    v = sp - s * (float)gold[idx];
  }
  red[t] = v;
  __syncthreads();
  for (int st = 128; st > 0; st >>= 1) {
    if (t < st) red[t] += red[t + st];
    __syncthreads();
  }
  if (t == 0) part[blockIdx.x] = red[0];
}

// ---------------- span features: (end-start) concat maxpool -> bf16 --------
__global__ void k_gold_repr(const float* __restrict__ reprs, const float* __restrict__ word_repr,
                            const int* __restrict__ sstart, const int* __restrict__ send,
                            bf16_t* __restrict__ gr) {
  int row = blockIdx.x;          // b*A + a  (512 rows)
  int b = row >> 6;
  int s = sstart[row], e = send[row];
  for (int d = threadIdx.x; d < DD; d += blockDim.x) {
    float diff = reprs[((long)b * LL + e) * DD + d] - reprs[((long)b * LL + s) * DD + d];
    float m = NEGV;
    for (int w = s; w < e; ++w)
      m = fmaxf(m, word_repr[((long)b * WW + w) * DD + d]);
    gr[(long)row * (2 * DD) + d]      = (bf16_t)diff;
    gr[(long)row * (2 * DD) + DD + d] = (bf16_t)m;
  }
}

// ---------------- LayerNorm + ReLU -> bf16 ----------------
__global__ void k_ln_relu(const float* __restrict__ h, const float* __restrict__ gamma,
                          const float* __restrict__ beta, bf16_t* __restrict__ out) {
  __shared__ float rs[256], rq[256];
  int row = blockIdx.x, t = threadIdx.x;       // 512 rows, E=256
  float x = h[(long)row * EE + t];
  rs[t] = x; rq[t] = x * x;
  __syncthreads();
  for (int st = 128; st > 0; st >>= 1) {
    if (t < st) { rs[t] += rs[t + st]; rq[t] += rq[t + st]; }
    __syncthreads();
  }
  float mu  = rs[0] / EE;
  float var = rq[0] / EE - mu * mu;
  float y = (x - mu) * rsqrtf(var + 1e-5f) * gamma[t] + beta[t];
  out[(long)row * EE + t] = (bf16_t)fmaxf(y, 0.0f);
}

// ---------------- per-row log-softmax cross entropy ----------------
__global__ void k_ce(const float* __restrict__ logits, const int* __restrict__ chart,
                     const int* __restrict__ action_len, float* __restrict__ part) {
  __shared__ float red[128];
  int row = blockIdx.x, t = threadIdx.x;       // 512 rows, V=128
  float x = logits[(long)row * VV + t];
  red[t] = x;
  __syncthreads();
  for (int st = 64; st > 0; st >>= 1) {
    if (t < st) red[t] = fmaxf(red[t], red[t + st]);
    __syncthreads();
  }
  float mx = red[0];
  __syncthreads();
  red[t] = expf(x - mx);
  __syncthreads();
  for (int st = 64; st > 0; st >>= 1) {
    if (t < st) red[t] += red[t + st];
    __syncthreads();
  }
  if (t == 0) {
    float lse = mx + logf(red[0]);
    int b = row >> 6, a = row & 63;
    int tgt = chart[row];
    float ce = lse - logits[(long)row * VV + tgt];
    part[row] = (a < action_len[b]) ? ce : 0.0f;
  }
}

// ---------------- final deterministic reduction ----------------
__global__ void k_final(const float* __restrict__ span_part, const float* __restrict__ ce_part,
                        const int* __restrict__ seq_len, const int* __restrict__ action_len,
                        float* __restrict__ out) {
  __shared__ float ra[256], rb[256];
  int t = threadIdx.x;
  float sa = 0.f, sb = 0.f;
  for (int i = t; i < 2048; i += 256) sa += span_part[i];
  for (int i = t; i < 512;  i += 256) sb += ce_part[i];
  ra[t] = sa; rb[t] = sb;
  __syncthreads();
  for (int st = 128; st > 0; st >>= 1) {
    if (t < st) { ra[t] += ra[t + st]; rb[t] += rb[t + st]; }
    __syncthreads();
  }
  if (t == 0) {
    float span_cnt = 0.f, lab_cnt = 0.f;
    for (int b = 0; b < BB; ++b) {
      float s = (float)seq_len[b];
      span_cnt += s * (s + 1.0f) * 0.5f;
      lab_cnt  += (float)action_len[b];
    }
    out[0] = ra[0] / span_cnt + rb[0] / lab_cnt;
  }
}

// ---------------- host side ----------------
extern "C" void kernel_launch(void* const* d_in, const int* in_sizes, int n_in,
                              void* d_out, int out_size, void* d_ws, size_t ws_size,
                              hipStream_t stream) {
  (void)in_sizes; (void)n_in; (void)out_size; (void)ws_size;
  const float* reprs      = (const float*)d_in[0];
  const float* word_repr  = (const float*)d_in[1];
  const int*   seq_len    = (const int*)d_in[2];
  const int*   gold_span  = (const int*)d_in[3];
  const int*   span_start = (const int*)d_in[4];
  const int*   span_end   = (const int*)d_in[5];
  const int*   chart      = (const int*)d_in[6];
  const int*   action_len = (const int*)d_in[7];
  const float* W_biaffine = (const float*)d_in[8];
  const float* Wp1        = (const float*)d_in[9];
  const float* bp1        = (const float*)d_in[10];
  const float* ln_gamma   = (const float*)d_in[11];
  const float* ln_beta    = (const float*)d_in[12];
  const float* Wp2        = (const float*)d_in[13];
  const float* bp2        = (const float*)d_in[14];
  const float* label_emb  = (const float*)d_in[15];
  float* out = (float*)d_out;

  // workspace carve-up (256B aligned)
  char* w = (char*)d_ws;
  auto alloc = [&](size_t bytes) {
    char* p = w;
    w += (bytes + 255) & ~(size_t)255;
    return p;
  };
  bf16_t* xb      = (bf16_t*)alloc((size_t)BB * LL * KPAD * 2);   // [B,256,1056]
  bf16_t* Wt      = (bf16_t*)alloc((size_t)NPAD * KPAD * 2);      // [1088,1056]
  bf16_t* T       = (bf16_t*)alloc((size_t)BB * LL * NPAD * 2);   // [B,256,1088]
  float*  S       = (float*) alloc((size_t)BB * LL * LL * 4);     // [B,256,256]
  unsigned char* gold = (unsigned char*)alloc((size_t)BB * LL * LL);
  bf16_t* grep    = (bf16_t*)alloc((size_t)BB * AA * 2 * DD * 2); // [512,2048]
  bf16_t* w1b     = (bf16_t*)alloc((size_t)EE * 2 * DD * 2);      // [256,2048]
  bf16_t* w2b     = (bf16_t*)alloc((size_t)EE * EE * 2);          // [256,256]
  bf16_t* labb    = (bf16_t*)alloc((size_t)VV * EE * 2);          // [128,256]
  float*  h1f     = (float*) alloc((size_t)BB * AA * EE * 4);     // [512,256]
  bf16_t* hbf     = (bf16_t*)alloc((size_t)BB * AA * EE * 2);
  bf16_t* h2b     = (bf16_t*)alloc((size_t)BB * AA * EE * 2);
  float*  logits  = (float*) alloc((size_t)BB * AA * VV * 4);     // [512,128]
  float*  span_pt = (float*) alloc(2048 * 4);
  float*  ce_pt   = (float*) alloc(512 * 4);

  // --- prep: bf16 conversions, padded operands, gold map ---
  {
    int n = BB * LL * KPAD;
    k_build_xb<<<(n + 255) / 256, 256, 0, stream>>>(reprs, xb, n);
  }
  {
    int n = NPAD * KPAD;
    k_build_wt<<<(n + 255) / 256, 256, 0, stream>>>(W_biaffine, Wt, n);
  }
  k_f32_to_bf16<<<(EE * 2 * DD + 255) / 256, 256, 0, stream>>>(Wp1, w1b, EE * 2 * DD);
  k_f32_to_bf16<<<(EE * EE + 255) / 256, 256, 0, stream>>>(Wp2, w2b, EE * EE);
  k_f32_to_bf16<<<(VV * EE + 255) / 256, 256, 0, stream>>>(label_emb, labb, VV * EE);
  k_zero_u32<<<(BB * LL * LL / 4 + 255) / 256, 256, 0, stream>>>((unsigned int*)gold, BB * LL * LL / 4);
  k_scatter_gold<<<(GG + 255) / 256, 256, 0, stream>>>(gold_span, gold);

  // --- GEMM1: T[b] = Xb[b] @ W     (M=256, N=1088, K=1056, bf16 out) ---
  k_gemm_bf16<<<dim3(NPAD / 64, LL / 64, BB), 128, 0, stream>>>(
      xb, KPAD, (long)LL * KPAD, Wt, KPAD, 0,
      T, NPAD, (long)LL * NPAD, nullptr, KPAD, 1);

  // --- GEMM2: S[b] = T[b] @ Xb[b]^T  (M=256, N=256, K=1056, f32 out) ---
  k_gemm_bf16<<<dim3(LL / 64, LL / 64, BB), 128, 0, stream>>>(
      T, NPAD, (long)LL * NPAD, xb, KPAD, (long)LL * KPAD,
      S, LL, (long)LL * LL, nullptr, KPAD, 0);

  // --- span BCE loss partials ---
  k_span_loss<<<2048, 256, 0, stream>>>(S, gold, seq_len, span_pt);

  // --- span features ---
  k_gold_repr<<<BB * AA, 256, 0, stream>>>(reprs, word_repr, span_start, span_end, grep);

  // --- GEMM3: h1 = gold_repr @ Wp1^T + bp1  (M=512, N=256, K=2048) ---
  k_gemm_bf16<<<dim3(EE / 64, BB * AA / 64, 1), 128, 0, stream>>>(
      grep, 2 * DD, 0, w1b, 2 * DD, 0,
      h1f, EE, 0, bp1, 2 * DD, 0);

  // --- LN + ReLU -> bf16 ---
  k_ln_relu<<<BB * AA, EE, 0, stream>>>(h1f, ln_gamma, ln_beta, hbf);

  // --- GEMM4: h2 = h @ Wp2^T + bp2  (M=512, N=256, K=256, bf16 out) ---
  k_gemm_bf16<<<dim3(EE / 64, BB * AA / 64, 1), 128, 0, stream>>>(
      hbf, EE, 0, w2b, EE, 0,
      h2b, EE, 0, bp2, EE, 1);

  // --- GEMM5: logits = h2 @ label_emb^T  (M=512, N=128, K=256) ---
  k_gemm_bf16<<<dim3(VV / 64, BB * AA / 64, 1), 128, 0, stream>>>(
      h2b, EE, 0, labb, EE, 0,
      logits, VV, 0, nullptr, EE, 0);

  // --- CE partials + final combine ---
  k_ce<<<BB * AA, VV, 0, stream>>>(logits, chart, action_len, ce_pt);
  k_final<<<1, 256, 0, stream>>>(span_pt, ce_pt, seq_len, action_len, out);
}
